// TransformerMoELayer_77558519431560
// MI455X (gfx1250) — matrix operand order repaired and checked by reading, not compiled, not run on hardware
//
#include <hip/hip_runtime.h>
#include <hip/hip_bf16.h>

// ---------------------------------------------------------------------------
// MoE layer for MI455X (gfx1250, wave32, WMMA 16x16x32 f16,
// GLOBAL_LOAD_ASYNC_TO_LDS double-buffered staging).
// T=8192 tokens, D=1024, F=4096, E=8 experts, top-2 routing.
// Pipeline: gate -> scan -> build -> x_to_f16 -> fc(+gelu) -> proj -> combine
// ---------------------------------------------------------------------------

#define TOKENS   8192
#define DDIM     1024
#define FDIM     4096
#define NEXP     8
#define TOPK     2
#define NROWS    (TOKENS * TOPK)   // 16384 compacted (token,slot) rows
#define ASTRIDE  40                // LDS row stride in halfs (16B aligned, conflict-spread)

typedef __attribute__((ext_vector_type(16))) _Float16 v16h;
typedef __attribute__((ext_vector_type(8)))  _Float16 v8h;
typedef __attribute__((ext_vector_type(4)))  _Float16 v4h;
typedef __attribute__((ext_vector_type(8)))  float    v8f;

// -------------------- CDNA5 async memory->LDS helpers -----------------------

// Low 32 bits of a flat shared-space address are the DS byte offset
// (ISA 10.2 aperture mapping).
__device__ __forceinline__ unsigned lds_byte_off(const void* p) {
  return (unsigned)(uintptr_t)p;
}

// Copy 32 bytes/lane memory->LDS via ASYNCcnt-tracked DMA (no VGPR data path).
// INST_OFFSET is added to BOTH the LDS and global address (ISA 10, async ops),
// so one base pair covers both 16B halves.
__device__ __forceinline__ void async_copy32(unsigned lds_dst, const void* gsrc) {
  asm volatile(
      "global_load_async_to_lds_b128 %0, %1, off\n\t"
      "global_load_async_to_lds_b128 %0, %1, off offset:16"
      :: "v"(lds_dst), "v"((unsigned long long)(uintptr_t)gsrc)
      : "memory");
}

__device__ __forceinline__ void wait_async0() {
  asm volatile("s_wait_asynccnt 0x0" ::: "memory");
}

// ---------------- WMMA fragment helpers (contiguous k order, same in A and B) ----------------

// A (16x32 f16): lane<16 -> M=lane, K = {0..15}; lane>=16 -> M=lane-16,
// K = {16..31}. Element j: k = b + j, b = (lane>>4)*16.
__device__ __forceinline__ v16h load_a_frag(const _Float16* As, int lane, int g) {
  const int m = lane & 15;
  const int b = (lane >> 4) * 16;
  const _Float16* rowp = As + (size_t)(g * 16 + m) * ASTRIDE;
  v8h lo = *(const v8h*)(rowp + b);        // k = b .. b+7
  v8h hi = *(const v8h*)(rowp + b + 8);    // k = b+8 .. b+15
  v16h a;
#pragma unroll
  for (int j = 0; j < 8; ++j) { a[j] = lo[j]; a[j + 8] = hi[j]; }
  return a;
}

// B (32x16 f16): lane holds column n = lane&15; k = (lane>>4)*16 + j, j=0..15.
// wrow points at w[col*ld + (lane>>4)*16 + kk]: 16 consecutive f32, cvt to f16.
__device__ __forceinline__ v16h load_b_frag_f32(const float* __restrict__ wrow) {
  v16h bf;
#pragma unroll
  for (int j = 0; j < 16; j += 4) {
    float4 f = *(const float4*)(wrow + j);
    bf[j + 0] = (_Float16)f.x;
    bf[j + 1] = (_Float16)f.y;
    bf[j + 2] = (_Float16)f.z;
    bf[j + 3] = (_Float16)f.w;
  }
  return bf;
}

// ------------------------------- routing ------------------------------------

__global__ void moe_zero(int* __restrict__ counts) {
  if (threadIdx.x < NEXP) counts[threadIdx.x] = 0;
}

// one wave per token: logits, top-2, softmax, rank assignment
__global__ __launch_bounds__(256) void moe_gate(
    const float* __restrict__ x, const float* __restrict__ gw,
    int* __restrict__ counts, int* __restrict__ idx,
    float* __restrict__ wts, int* __restrict__ rnk) {
  const int gtid = blockIdx.x * blockDim.x + threadIdx.x;
  const int t    = gtid >> 5;
  const int lane = threadIdx.x & 31;
  if (t >= TOKENS) return;

  float acc[NEXP];
#pragma unroll
  for (int e = 0; e < NEXP; ++e) acc[e] = 0.f;

  const float* xr = x + (size_t)t * DDIM;
  for (int d = lane; d < DDIM; d += 32) {
    const float xv = xr[d];
#pragma unroll
    for (int e = 0; e < NEXP; ++e) acc[e] += xv * gw[e * DDIM + d];
  }
#pragma unroll
  for (int off = 16; off > 0; off >>= 1) {
#pragma unroll
    for (int e = 0; e < NEXP; ++e) acc[e] += __shfl_down(acc[e], off, 32);
  }
  if (lane == 0) {
    int i0 = 0; float v0 = acc[0];
#pragma unroll
    for (int e = 1; e < NEXP; ++e)
      if (acc[e] > v0) { v0 = acc[e]; i0 = e; }
    int i1 = -1; float v1 = -3.4e38f;
#pragma unroll
    for (int e = 0; e < NEXP; ++e)
      if (e != i0 && acc[e] > v1) { v1 = acc[e]; i1 = e; }
    const float w0 = 1.f / (1.f + __expf(v1 - v0));
    idx[t * 2 + 0] = i0;  idx[t * 2 + 1] = i1;
    wts[t * 2 + 0] = w0;  wts[t * 2 + 1] = 1.f - w0;
    rnk[t * 2 + 0] = atomicAdd(&counts[i0], 1);
    rnk[t * 2 + 1] = atomicAdd(&counts[i1], 1);
  }
}

__global__ void moe_scan(const int* __restrict__ counts, int* __restrict__ offsets) {
  if (threadIdx.x == 0 && blockIdx.x == 0) {
    int s = 0;
#pragma unroll
    for (int e = 0; e < NEXP; ++e) { offsets[e] = s; s += counts[e]; }
  }
}

__global__ void moe_build(const int* __restrict__ idx, const int* __restrict__ rnk,
                          const int* __restrict__ offsets,
                          int* __restrict__ rows, int* __restrict__ token_list) {
  const int t = blockIdx.x * blockDim.x + threadIdx.x;
  if (t >= TOKENS) return;
#pragma unroll
  for (int s = 0; s < TOPK; ++s) {
    const int e = idx[t * 2 + s];
    const int r = offsets[e] + rnk[t * 2 + s];
    rows[t * 2 + s] = r;
    token_list[r]   = t;
  }
}

__global__ void x_to_f16(const float* __restrict__ x, _Float16* __restrict__ Xh) {
  const size_t i = (size_t)blockIdx.x * blockDim.x + threadIdx.x;  // T*D/4
  float4 f = *(const float4*)(x + i * 4);
  v4h h;
  h[0] = (_Float16)f.x; h[1] = (_Float16)f.y;
  h[2] = (_Float16)f.z; h[3] = (_Float16)f.w;
  *(v4h*)(Xh + i * 4) = h;
}

// ----------------------- pass A: H = gelu(Xg @ Wfc^T + b) -------------------
// grid (F/128, 64, E), block 256 = 8 waves; wave w owns cols [w*16, w*16+16)

__global__ __launch_bounds__(256) void moe_fc(
    const _Float16* __restrict__ Xh, const float* __restrict__ w_fc,
    const float* __restrict__ b_fc, const int* __restrict__ counts,
    const int* __restrict__ offsets, const int* __restrict__ token_list,
    _Float16* __restrict__ H) {
  const int e = blockIdx.z;
  const int cnt = counts[e];
  const int rowTile = blockIdx.y;
  if (rowTile * 128 >= cnt) return;
  const int base    = offsets[e];
  const int colBase = blockIdx.x * 128;
  const int tid  = threadIdx.x;
  const int lane = tid & 31;
  const int wave = tid >> 5;

  __shared__ _Float16 As[2][128 * ASTRIDE];
  __shared__ int toks[128];

  if (tid < 128) {
    int r = base + rowTile * 128 + tid;
    r = min(r, NROWS - 1);                 // clamp tile overrun (garbage rows masked at store)
    toks[tid] = token_list[r];
  }
  __syncthreads();

  v8f c[8];
#pragma unroll
  for (int g = 0; g < 8; ++g)
#pragma unroll
    for (int r = 0; r < 8; ++r) c[g][r] = 0.f;

  const int col = colBase + wave * 16 + (lane & 15);
  const float* wcol = w_fc + ((size_t)e * FDIM + col) * DDIM + ((lane >> 4) * 16);

  // per-thread async staging: 32B/lane, 128 rows x 32 halfs per buffer
  const int srow  = tid >> 1;
  const int shalf = (tid & 1) * 16;
  const _Float16* gsrc = Xh + (size_t)toks[srow] * DDIM + shalf;
  const unsigned ldst0 = lds_byte_off(&As[0][srow * ASTRIDE + shalf]);
  const unsigned ldst1 = lds_byte_off(&As[1][srow * ASTRIDE + shalf]);

  async_copy32(ldst0, gsrc);               // prologue: buffer 0 <- k=0 slice
  wait_async0();
  __syncthreads();

  int cur = 0;
  for (int kk = 0; kk < DDIM; kk += 32) {
    if (kk + 32 < DDIM) {
      async_copy32(cur ? ldst0 : ldst1, gsrc + kk + 32);   // prefetch next slice
      __builtin_prefetch(wcol + kk + 32, 0, 3);            // global_prefetch_b8
    }
    const v16h b = load_b_frag_f32(wcol + kk);
#pragma unroll
    for (int g = 0; g < 8; ++g) {
      const v16h a = load_a_frag(As[cur], lane, g);
      c[g] = __builtin_amdgcn_wmma_f32_16x16x32_f16(false, a, false, b,
                                                    (short)0, c[g], false, false);
    }
    wait_async0();                          // our transfers landed in LDS
    __syncthreads();                        // everyone's transfers visible
    cur ^= 1;
  }

  const float bias = b_fc[e * FDIM + col];
  const int rmax = cnt - rowTile * 128;
#pragma unroll
  for (int g = 0; g < 8; ++g) {
#pragma unroll
    for (int r = 0; r < 8; ++r) {
      const int m = r + 8 * (lane >> 4);
      const int local = g * 16 + m;
      if (local < rmax) {
        const float v = c[g][r] + bias;
        const float th = tanhf(0.7978845608028654f * (v + 0.044715f * v * v * v));
        const float gl = 0.5f * v * (1.0f + th);
        H[(size_t)(base + rowTile * 128 + local) * FDIM + col] = (_Float16)gl;
      }
    }
  }
}

// ----------------------- pass B: Y = H @ Wproj^T + b ------------------------
// grid (D/128, 64, E); A rows are compact (no gather)

__global__ __launch_bounds__(256) void moe_proj(
    const _Float16* __restrict__ H, const float* __restrict__ w_proj,
    const float* __restrict__ b_proj, const int* __restrict__ counts,
    const int* __restrict__ offsets, float* __restrict__ Y) {
  const int e = blockIdx.z;
  const int cnt = counts[e];
  const int rowTile = blockIdx.y;
  if (rowTile * 128 >= cnt) return;
  const int base    = offsets[e];
  const int colBase = blockIdx.x * 128;
  const int tid  = threadIdx.x;
  const int lane = tid & 31;
  const int wave = tid >> 5;

  __shared__ _Float16 As[2][128 * ASTRIDE];

  v8f c[8];
#pragma unroll
  for (int g = 0; g < 8; ++g)
#pragma unroll
    for (int r = 0; r < 8; ++r) c[g][r] = 0.f;

  const int col = colBase + wave * 16 + (lane & 15);
  const float* wcol = w_proj + ((size_t)e * DDIM + col) * FDIM + ((lane >> 4) * 16);

  const int srow  = tid >> 1;
  const int shalf = (tid & 1) * 16;
  const int arow  = min(base + rowTile * 128 + srow, NROWS - 1);  // clamp overrun
  const _Float16* gsrc = H + (size_t)arow * FDIM + shalf;
  const unsigned ldst0 = lds_byte_off(&As[0][srow * ASTRIDE + shalf]);
  const unsigned ldst1 = lds_byte_off(&As[1][srow * ASTRIDE + shalf]);

  async_copy32(ldst0, gsrc);
  wait_async0();
  __syncthreads();

  int cur = 0;
  for (int kk = 0; kk < FDIM; kk += 32) {
    if (kk + 32 < FDIM) {
      async_copy32(cur ? ldst0 : ldst1, gsrc + kk + 32);
      __builtin_prefetch(wcol + kk + 32, 0, 3);
    }
    const v16h b = load_b_frag_f32(wcol + kk);
#pragma unroll
    for (int g = 0; g < 8; ++g) {
      const v16h a = load_a_frag(As[cur], lane, g);
      c[g] = __builtin_amdgcn_wmma_f32_16x16x32_f16(false, a, false, b,
                                                    (short)0, c[g], false, false);
    }
    wait_async0();
    __syncthreads();
    cur ^= 1;
  }

  const float bias = b_proj[e * DDIM + col];
  const int rmax = cnt - rowTile * 128;
#pragma unroll
  for (int g = 0; g < 8; ++g) {
#pragma unroll
    for (int r = 0; r < 8; ++r) {
      const int m = r + 8 * (lane >> 4);
      const int local = g * 16 + m;
      if (local < rmax)
        Y[(size_t)(base + rowTile * 128 + local) * DDIM + col] = c[g][r] + bias;
    }
  }
}

// ------------------- combine: out[t] = w0*Y[r0] + w1*Y[r1] ------------------

__global__ void moe_combine(const float* __restrict__ Y, const int* __restrict__ rows,
                            const float* __restrict__ wts, float* __restrict__ out) {
  const size_t i = (size_t)blockIdx.x * blockDim.x + threadIdx.x;   // T*D/4
  const int t  = (int)(i / (DDIM / 4));
  const int d4 = (int)(i % (DDIM / 4));
  const int r0 = rows[t * 2 + 0], r1 = rows[t * 2 + 1];
  const float w0 = wts[t * 2 + 0], w1 = wts[t * 2 + 1];
  const float4 y0 = *(const float4*)(Y + (size_t)r0 * DDIM + d4 * 4);
  const float4 y1 = *(const float4*)(Y + (size_t)r1 * DDIM + d4 * 4);
  float4 o;
  o.x = w0 * y0.x + w1 * y1.x;
  o.y = w0 * y0.y + w1 * y1.y;
  o.z = w0 * y0.z + w1 * y1.z;
  o.w = w0 * y0.w + w1 * y1.w;
  *(float4*)(out + (size_t)t * DDIM + d4 * 4) = o;
}

// ------------------------------- host launch --------------------------------

static inline char* ws_take(char*& p, size_t bytes) {
  char* r = p;
  p += (bytes + 255) & ~(size_t)255;
  return r;
}

extern "C" void kernel_launch(void* const* d_in, const int* in_sizes, int n_in,
                              void* d_out, int out_size, void* d_ws, size_t ws_size,
                              hipStream_t stream) {
  (void)in_sizes; (void)n_in; (void)out_size; (void)ws_size;
  const float* x      = (const float*)d_in[0];   // [4,2048,1024]
  const float* gate_w = (const float*)d_in[1];   // [E,D]
  const float* w_fc   = (const float*)d_in[2];   // [E,F,D]
  const float* b_fc   = (const float*)d_in[3];   // [E,F]
  const float* w_proj = (const float*)d_in[4];   // [E,D,F]
  const float* b_proj = (const float*)d_in[5];   // [E,D]
  float* out = (float*)d_out;

  char* p = (char*)d_ws;
  int*      counts     = (int*)ws_take(p, NEXP * sizeof(int));
  int*      offsets    = (int*)ws_take(p, NEXP * sizeof(int));
  int*      idx        = (int*)ws_take(p, (size_t)TOKENS * 2 * sizeof(int));
  float*    wts        = (float*)ws_take(p, (size_t)TOKENS * 2 * sizeof(float));
  int*      rnk        = (int*)ws_take(p, (size_t)TOKENS * 2 * sizeof(int));
  int*      rows       = (int*)ws_take(p, (size_t)TOKENS * 2 * sizeof(int));
  int*      token_list = (int*)ws_take(p, (size_t)NROWS * sizeof(int));
  _Float16* Xh         = (_Float16*)ws_take(p, (size_t)TOKENS * DDIM * sizeof(_Float16));
  float*    Y          = (float*)ws_take(p, (size_t)NROWS * DDIM * sizeof(float));
  _Float16* H          = (_Float16*)ws_take(p, (size_t)NROWS * FDIM * sizeof(_Float16));

  moe_zero<<<1, 32, 0, stream>>>(counts);
  moe_gate<<<TOKENS / 8, 256, 0, stream>>>(x, gate_w, counts, idx, wts, rnk);
  moe_scan<<<1, 32, 0, stream>>>(counts, offsets);
  moe_build<<<TOKENS / 256, 256, 0, stream>>>(idx, rnk, offsets, rows, token_list);
  x_to_f16<<<(TOKENS * DDIM / 4) / 256, 256, 0, stream>>>(x, Xh);

  dim3 gfc(FDIM / 128, TOKENS / 128, NEXP);
  moe_fc<<<gfc, 256, 0, stream>>>(Xh, w_fc, b_fc, counts, offsets, token_list, H);

  dim3 gpr(DDIM / 128, TOKENS / 128, NEXP);
  moe_proj<<<gpr, 256, 0, stream>>>(H, w_proj, b_proj, counts, offsets, Y);

  moe_combine<<<(TOKENS * DDIM / 4) / 256, 256, 0, stream>>>(Y, rows, wts, out);
}
